// HeteroGCLSTM_GAT_75642964017917
// MI455X (gfx1250) — compile-verified, matrix operand order checked by reading, not run on hardware
//
#include <hip/hip_runtime.h>
#include <hip/hip_bf16.h>

typedef __attribute__((ext_vector_type(16))) _Float16 v16h;
typedef __attribute__((ext_vector_type(8)))  float    v8f;

#define D 64
#define NEG_SLOPE 0.2f

// ---------- helpers ----------
__device__ __forceinline__ unsigned enc_f32(float f) {
    unsigned u = __float_as_uint(f);
    return (u & 0x80000000u) ? ~u : (u | 0x80000000u);
}
__device__ __forceinline__ float dec_f32(unsigned u) {
    u = (u & 0x80000000u) ? (u ^ 0x80000000u) : ~u;
    return __uint_as_float(u);
}
__device__ __forceinline__ float sigmoidf(float x) { return 1.0f / (1.0f + __expf(-x)); }

// ---------- 1) v_s[g] = Wsrc[g] @ att_src[g], v_d[g] = Wdst[g] @ att_dst[g] ----------
__global__ void prep_vecs(const float* __restrict__ Wsrc, const float* __restrict__ Wdst,
                          const float* __restrict__ att_src, const float* __restrict__ att_dst,
                          float* __restrict__ vs, float* __restrict__ vd) {
    int t = blockIdx.x * blockDim.x + threadIdx.x;   // 0..511
    if (t >= 2 * 4 * D) return;
    int sd = t >> 8;            // 0: src, 1: dst
    int g  = (t >> 6) & 3;
    int k  = t & 63;
    const float* W = (sd == 0 ? Wsrc : Wdst) + (size_t)g * D * D + (size_t)k * D;
    const float* a = (sd == 0 ? att_src : att_dst) + (size_t)g * D;
    float s = 0.f;
#pragma unroll
    for (int j = 0; j < D; ++j) s += W[j] * a[j];
    (sd == 0 ? vs : vd)[g * D + k] = s;
}

// ---------- 1b) pre-pack B fragments into per-lane WMMA layout (f16) ----------
// For mat 0..7 (0..3: W[g], 4..7: Wsrc[g]), col-tile ct 0..3, lane 0..31:
// 32 halfs per lane = fragments b0 (k 0..31) and b1 (k 32..63).
// ISA 7.12.2 16-bit B layout: lane -> column n = ct*16 + (lane&15), half = lane>>4,
// VGPR r holds K pair at kb = blk*32 + ((r&4)?16:0) + half*8 + (r&3)*2.
__global__ void pack_B(const float* __restrict__ W4, const float* __restrict__ Wsrc4,
                       _Float16* __restrict__ bpack) {
    int t = blockIdx.x * blockDim.x + threadIdx.x;   // 0..1023
    if (t >= 8 * 4 * 32) return;
    int mat  = t >> 7;
    int ct   = (t >> 5) & 3;
    int lane = t & 31;
    int g    = mat & 3;
    const float* B = (mat < 4 ? W4 : Wsrc4) + (size_t)g * D * D;
    int n    = ct * 16 + (lane & 15);
    int half = lane >> 4;
    _Float16* out = bpack + (size_t)t * 32;
#pragma unroll
    for (int blk = 0; blk < 2; ++blk) {
#pragma unroll
        for (int r = 0; r < 8; ++r) {
            int kb = blk * 32 + ((r & 4) ? 16 : 0) + (half << 3) + ((r & 3) << 1);
            out[blk * 16 + 2 * r]     = (_Float16)B[(size_t)kb * D + n];
            out[blk * 16 + 2 * r + 1] = (_Float16)B[(size_t)(kb + 1) * D + n];
        }
    }
}

// ---------- 2) alpha_s[g][n] = h[n]·vs[g], alpha_d[g][n] = h[n]·vd[g] ----------
__global__ void alpha_kernel(const float* __restrict__ h, const float* __restrict__ vs,
                             const float* __restrict__ vd, float* __restrict__ as,
                             float* __restrict__ ad, int N) {
    int t = blockIdx.x * blockDim.x + threadIdx.x;
    if (t >= 4 * N) return;
    int g = t / N, n = t % N;
    const float4* hr = (const float4*)(h + (size_t)n * D);
    const float4* s4 = (const float4*)(vs + g * D);
    const float4* d4 = (const float4*)(vd + g * D);
    float a = 0.f, bsum = 0.f;
#pragma unroll
    for (int j = 0; j < D / 4; ++j) {
        float4 hv = hr[j], sv = s4[j], dv = d4[j];
        a    += hv.x * sv.x + hv.y * sv.y + hv.z * sv.z + hv.w * sv.w;
        bsum += hv.x * dv.x + hv.y * dv.y + hv.z * dv.z + hv.w * dv.w;
    }
    as[t] = a;
    ad[t] = bsum;
}

// ---------- 3) WMMA GEMMs: pre[g] = x@W[g] + b[g] + gat_b[g]  (mat 0..3)
//                           hs[g]  = h@Wsrc[g]                  (mat 4..7)
// One wave = one 16x16 output tile; K=64 via two v_wmma_f32_16x16x32_f16.
// A frag: 4 runs of 8 contiguous floats at arow + blk*16 + half*8 -> f16 pairs.
// B frag: pre-packed by pack_B, two 32-byte per-lane reads.
__global__ void gemm_wmma(const float* __restrict__ X, const float* __restrict__ H,
                          const _Float16* __restrict__ bpack,
                          const float* __restrict__ bias, const float* __restrict__ gat_b,
                          float* __restrict__ pre, float* __restrict__ hs, int N) {
    const int lane = threadIdx.x & 31;
    const int wave = threadIdx.x >> 5;       // 0..3 -> column tile
    const int m    = lane & 15;
    const int half = lane >> 4;

    const int row0 = blockIdx.x * 16;
    const int col0 = wave * 16;
    const int mat  = blockIdx.y;             // 0..7
    const int g    = mat & 3;
    const bool isX = mat < 4;

    const float* A = isX ? X : H;
    int row = row0 + m;
    int rclamp = row < N ? row : N - 1;
    const float* arow = A + (size_t)rclamp * D;
    const int n = col0 + m;

    // ---- A fragments (f32 -> f16) ----
    v16h a0, a1;
    const float4* ar4 = (const float4*)arow;
#pragma unroll
    for (int blk = 0; blk < 4; ++blk) {
        float4 lo = ar4[blk * 4 + 2 * half];
        float4 hi = ar4[blk * 4 + 2 * half + 1];
        v16h& dstf = (blk < 2) ? a0 : a1;
        int o = (blk & 1) * 8;
        dstf[o + 0] = (_Float16)lo.x; dstf[o + 1] = (_Float16)lo.y;
        dstf[o + 2] = (_Float16)lo.z; dstf[o + 3] = (_Float16)lo.w;
        dstf[o + 4] = (_Float16)hi.x; dstf[o + 5] = (_Float16)hi.y;
        dstf[o + 6] = (_Float16)hi.z; dstf[o + 7] = (_Float16)hi.w;
    }

    // ---- B fragments (pre-packed) ----
    const v16h* bp = (const v16h*)(bpack + (((size_t)mat * 4 + wave) * 32 + lane) * 32);
    v16h b0 = bp[0];
    v16h b1 = bp[1];

    v8f c = {};
    c = __builtin_amdgcn_wmma_f32_16x16x32_f16(false, a0, false, b0, (short)0, c, false, false);
    c = __builtin_amdgcn_wmma_f32_16x16x32_f16(false, a1, false, b1, (short)0, c, false, false);

    // D layout: VGPR r -> M = r + 8*half, N = lane&15
    float add = isX ? (bias[g * D + n] + gat_b[g * D + n]) : 0.0f;
    float* outbase = (isX ? pre : hs) + ((size_t)g * N + row0 + 8 * half) * D + n;
    if (row0 + 16 <= N) {        // tile-uniform fast path: straight-line stores
#pragma unroll
        for (int r = 0; r < 8; ++r)
            outbase[(size_t)r * D] = c[r] + add;
    } else {
#pragma unroll
        for (int r = 0; r < 8; ++r)
            if (row0 + 8 * half + r < N)
                outbase[(size_t)r * D] = c[r] + add;
    }
}

// ---------- 4) per-edge logits + segment max ----------
__global__ void edge_pass1(const int* __restrict__ src, const int* __restrict__ dst,
                           const float* __restrict__ as, const float* __restrict__ ad,
                           float* __restrict__ ebuf, unsigned* __restrict__ maxw,
                           int N, int E) {
    int e = blockIdx.x * blockDim.x + threadIdx.x;
    int g = blockIdx.y;
    if (e >= E) return;
    int s = src[e], d = dst[e];
    float v = as[(size_t)g * N + s] + ad[(size_t)g * N + d];
    v = v > 0.f ? v : NEG_SLOPE * v;
    ebuf[(size_t)g * E + e] = v;
    atomicMax(&maxw[(size_t)g * N + d], enc_f32(v));
}

// ---------- 5) exp(e - max) + segment sum ----------
__global__ void edge_pass2(const int* __restrict__ dst, float* __restrict__ ebuf,
                           const unsigned* __restrict__ maxw, float* __restrict__ denom,
                           int N, int E) {
    int e = blockIdx.x * blockDim.x + threadIdx.x;
    int g = blockIdx.y;
    if (e >= E) return;
    int d = dst[e];
    float mx = dec_f32(maxw[(size_t)g * N + d]);
    float ex = __expf(ebuf[(size_t)g * E + e] - mx);
    ebuf[(size_t)g * E + e] = ex;
    unsafeAtomicAdd(&denom[(size_t)g * N + d], ex);
}

// ---------- 6) weighted scatter: pre[dst] += w * hs[src]; one wave per (edge,gate) ----------
__global__ void edge_pass3(const int* __restrict__ src, const int* __restrict__ dst,
                           const float* __restrict__ ebuf, const float* __restrict__ denom,
                           const float* __restrict__ hs, float* __restrict__ pre,
                           int N, int E) {
    int lane = threadIdx.x & 31;
    int wav  = threadIdx.x >> 5;
    int e    = blockIdx.x * 8 + wav;
    int g    = blockIdx.y;
    if (e >= E) return;
    int s = src[e], d = dst[e];
    float w = ebuf[(size_t)g * E + e] / denom[(size_t)g * N + d];
    const float2* hrow = (const float2*)(hs + ((size_t)g * N + s) * D);
    float2 v = hrow[lane];
    float* out = pre + ((size_t)g * N + d) * D + 2 * lane;
    unsafeAtomicAdd(out,     w * v.x);
    unsafeAtomicAdd(out + 1, w * v.y);
}

// ---------- 7) pointwise LSTM combine ----------
__global__ void combine(const float* __restrict__ pre, const float* __restrict__ c_old,
                        float* __restrict__ out, int N) {
    size_t idx = (size_t)blockIdx.x * blockDim.x + threadIdx.x;
    size_t tot = (size_t)N * D;
    if (idx >= tot) return;
    float i = sigmoidf(pre[idx]);
    float f = sigmoidf(pre[tot + idx]);
    float t = tanhf(pre[2 * tot + idx]);
    float o = sigmoidf(pre[3 * tot + idx]);
    float cn = f * c_old[idx] + i * t;
    float hn = o * tanhf(cn);
    out[idx]       = hn;   // h_new
    out[tot + idx] = cn;   // c_new
}

extern "C" void kernel_launch(void* const* d_in, const int* in_sizes, int n_in,
                              void* d_out, int out_size, void* d_ws, size_t ws_size,
                              hipStream_t stream) {
    const float* x        = (const float*)d_in[0];
    const int*   edge     = (const int*)d_in[1];
    const float* h        = (const float*)d_in[2];
    const float* c_old    = (const float*)d_in[3];
    const float* W4       = (const float*)d_in[4];
    const float* bias     = (const float*)d_in[5];
    const float* Wsrc     = (const float*)d_in[6];
    const float* Wdst     = (const float*)d_in[7];
    const float* att_src  = (const float*)d_in[8];
    const float* att_dst  = (const float*)d_in[9];
    const float* gat_b    = (const float*)d_in[10];

    const int N = in_sizes[0] / D;
    const int E = in_sizes[1] / 2;
    const int* src = edge;
    const int* dst = edge + E;

    // workspace layout (floats)
    float* pre   = (float*)d_ws;                         // 4*N*D
    float* hs    = pre + (size_t)4 * N * D;              // 4*N*D
    float* as    = hs  + (size_t)4 * N * D;              // 4*N
    float* ad    = as  + (size_t)4 * N;                  // 4*N
    float* denom = ad  + (size_t)4 * N;                  // 4*N
    unsigned* maxw = (unsigned*)(denom + (size_t)4 * N); // 4*N (contiguous after denom)
    float* ebuf  = (float*)(maxw + (size_t)4 * N);       // 4*E
    float* vs    = ebuf + (size_t)4 * E;                 // 4*D
    float* vd    = vs + 4 * D;                           // 4*D
    _Float16* bpack = (_Float16*)(vd + 4 * D);           // 8*4*32*32 halfs = 64 KB

    // zero denom + maxw in one contiguous memset (graph-capturable)
    hipMemsetAsync(denom, 0, (size_t)8 * N * sizeof(float), stream);

    prep_vecs<<<2, 256, 0, stream>>>(Wsrc, Wdst, att_src, att_dst, vs, vd);
    pack_B<<<4, 256, 0, stream>>>(W4, Wsrc, bpack);

    alpha_kernel<<<(4 * N + 255) / 256, 256, 0, stream>>>(h, vs, vd, as, ad, N);

    {
        dim3 grid((N + 15) / 16, 8);
        gemm_wmma<<<grid, 128, 0, stream>>>(x, h, bpack, bias, gat_b, pre, hs, N);
    }

    {
        dim3 grid((E + 255) / 256, 4);
        edge_pass1<<<grid, 256, 0, stream>>>(src, dst, as, ad, ebuf, maxw, N, E);
        edge_pass2<<<grid, 256, 0, stream>>>(dst, ebuf, maxw, denom, N, E);
    }

    {
        dim3 grid((E + 7) / 8, 4);
        edge_pass3<<<grid, 256, 0, stream>>>(src, dst, ebuf, denom, hs, pre, N, E);
    }

    combine<<<((size_t)N * D + 255) / 256, 256, 0, stream>>>(pre, c_old, (float*)d_out, N);
}
